// MultiHeadAttention_4002909520603
// MI455X (gfx1250) — matrix operand order repaired and checked by reading, not compiled, not run on hardware
//
#include <hip/hip_runtime.h>
#include <hip/hip_bf16.h>

// ---------------------------------------------------------------------------
// MI455X (gfx1250) MHA forward.
//   d_in: x[2,2048,1024] f32, Wqkv[1024,3072] f32, bqkv[3072] f32,
//         Wo[1024,1024] f32, bo[1024] f32
//   d_out: out[2,2048,1024] f32  ++  attention[2,16,2048,2048] f32
//   ws: qkv bf16 (24MB) ++ vals bf16 (8MB)
// Matrix math: v_wmma_f32_16x16x32_bf16 (wave32).
// V staging: global_load_async_to_lds_b128 + s_wait_asynccnt (ASYNCcnt path).
// ---------------------------------------------------------------------------

typedef __attribute__((ext_vector_type(16))) __bf16 v16bf;
typedef __attribute__((ext_vector_type(8)))  __bf16 v8bf;
typedef __attribute__((ext_vector_type(8)))  float  v8f;

#define S_LEN 2048
#define D_MODEL 1024
#define HD 64
#define NH 16
#define QKV_ROW 3072
#define BQ 16

static __device__ __forceinline__ v8f wmma_bf16(v16bf a, v16bf b, v8f c) {
  return __builtin_amdgcn_wmma_f32_16x16x32_bf16(false, a, false, b,
                                                 (short)0, c, false, false);
}

// Async copy of 16 bytes global -> LDS (per active lane), tracked by ASYNCcnt.
static __device__ __forceinline__ void async_copy_b128(const void* gsrc, void* lds_dst) {
  unsigned long long ga = (unsigned long long)gsrc;
  unsigned lo = (unsigned)(unsigned long long)lds_dst;   // low 32 bits = LDS offset
  asm volatile("global_load_async_to_lds_b128 %0, %1, off"
               :: "v"(lo), "v"(ga) : "memory");
}
static __device__ __forceinline__ void wait_async0() {
  asm volatile("s_wait_asynccnt 0x0" ::: "memory");
}

// ---------------------------------------------------------------------------
// Generic tiled GEMM with bias:  out[M,N] = A[M,K] @ W[K,N] + bias[N]
// Block: 256 threads (8 waves), tile 64(M) x 128(N), K step 32.
// ---------------------------------------------------------------------------
template <typename AT, typename OT>
__global__ __launch_bounds__(256) void gemm_bias(const AT* __restrict__ A,
                                                 const float* __restrict__ W,
                                                 const float* __restrict__ bias,
                                                 OT* __restrict__ out,
                                                 int M, int N, int K) {
  __shared__ __bf16 As[64][32];    // 4 KB, row-major [m][k]
  __shared__ __bf16 Bt[128][32];   // 8 KB, N-major [n][k] (B operand layout)

  const int tid  = threadIdx.x;
  const int lane = tid & 31;
  const int wid  = tid >> 5;
  const int l16  = lane & 15;
  const int half = lane >> 4;
  const int waveM = wid >> 2;
  const int waveN = wid & 3;
  const int mBase = blockIdx.y * 64;
  const int nBase = blockIdx.x * 128;

  v8f c[2][2] = {};

  for (int k0 = 0; k0 < K; k0 += 32) {
    {
      int r  = tid >> 2;
      int cb = (tid & 3) * 8;
      const AT* src = A + (size_t)(mBase + r) * K + k0 + cb;
#pragma unroll
      for (int j = 0; j < 8; ++j) As[r][cb + j] = (__bf16)(float)src[j];
    }
    {
      int kr = tid >> 3;
      int nb = (tid & 7) * 16;
      const float* src = W + (size_t)(k0 + kr) * N + nBase + nb;
#pragma unroll
      for (int j = 0; j < 16; ++j) Bt[nb + j][kr] = (__bf16)src[j];
    }
    __syncthreads();

    v16bf afrag[2], bfrag[2];
#pragma unroll
    for (int mf = 0; mf < 2; ++mf) {
      int row = waveM * 32 + mf * 16 + l16;
      int kb  = half * 16;
      v8bf lo = *(const v8bf*)&As[row][kb];
      v8bf hi = *(const v8bf*)&As[row][kb + 8];
#pragma unroll
      for (int i = 0; i < 8; ++i) { afrag[mf][i] = lo[i]; afrag[mf][i + 8] = hi[i]; }
    }
#pragma unroll
    for (int nf = 0; nf < 2; ++nf) {
      int ncol = waveN * 32 + nf * 16 + l16;
      bfrag[nf] = *(const v16bf*)&Bt[ncol][half * 16];
    }
#pragma unroll
    for (int mf = 0; mf < 2; ++mf)
#pragma unroll
      for (int nf = 0; nf < 2; ++nf)
        c[mf][nf] = wmma_bf16(afrag[mf], bfrag[nf], c[mf][nf]);
    __syncthreads();
  }

#pragma unroll
  for (int mf = 0; mf < 2; ++mf)
#pragma unroll
    for (int nf = 0; nf < 2; ++nf) {
      int col = nBase + waveN * 32 + nf * 16 + l16;
      float bv = bias[col];
#pragma unroll
      for (int i = 0; i < 8; ++i) {
        int row = mBase + waveM * 32 + mf * 16 + (half ? i + 8 : i);
        out[(size_t)row * N + col] = (OT)(c[mf][nf][i] + bv);
      }
    }
}

// ---------------------------------------------------------------------------
// Fused attention: one block = (b, h, 16 query rows).
// Phase 1: logits tile [16 x 2048] via WMMA -> LDS (fp32).
// Phase 2: row softmax, single HBM write of attention.
// Phase 3: values = A @ V, V staged via async LDS copies.
// LDS: Qs 2KB + Lf 128KB + red3 4KB + Vs 8KB = ~142KB (< 320KB/WGP).
// ---------------------------------------------------------------------------
__global__ __launch_bounds__(256) void attn_kernel(const __bf16* __restrict__ qkvb,
                                                   float* __restrict__ att,
                                                   __bf16* __restrict__ vals) {
  __shared__ __bf16 Qs[BQ][HD];
  __shared__ float  Lf[BQ][S_LEN];
  __shared__ float  red3[4][16][16];
  __shared__ __bf16 Vs[64][64];        // V super-chunk [k][n], 8 KB

  const int tid  = threadIdx.x;
  const int lane = tid & 31;
  const int wid  = tid >> 5;
  const int l16  = lane & 15;
  const int half = lane >> 4;
  const int bh   = blockIdx.y;
  const int b    = bh >> 4;
  const int hh   = bh & 15;
  const int qBase = blockIdx.x * BQ;
  const size_t rowOff = (size_t)b * S_LEN * QKV_ROW;

  // --- load Q tile [16 x 64] (q slice = h*192 + 0..63) ---
  {
    int r  = tid >> 4;
    int d0 = (tid & 15) * 4;
    const __bf16* src = qkvb + rowOff + (size_t)(qBase + r) * QKV_ROW + hh * 192 + d0;
    *(unsigned long long*)&Qs[r][d0] = *(const unsigned long long*)src;
  }
  __syncthreads();

  // ---------------- Phase 1: logits = scale * Q K^T ----------------
  {
    const float scale = 0.125f;        // 1/sqrt(64)
    v16bf aq[2];
#pragma unroll
    for (int kc = 0; kc < 2; ++kc) {
      int kb = half * 16;
      v8bf lo = *(const v8bf*)&Qs[l16][kc * 32 + kb];
      v8bf hi = *(const v8bf*)&Qs[l16][kc * 32 + 8 + kb];
#pragma unroll
      for (int i = 0; i < 8; ++i) { aq[kc][i] = lo[i]; aq[kc][i + 8] = hi[i]; }
    }
    for (int ct = 0; ct < 16; ++ct) {
      int colBase = wid * 256 + ct * 16;
      int col = colBase + l16;
      const __bf16* krow = qkvb + rowOff + (size_t)col * QKV_ROW + hh * 192 + 64;
      v8f acc = {};
#pragma unroll
      for (int kc = 0; kc < 2; ++kc) {
        v16bf bk = *(const v16bf*)(krow + kc * 32 + half * 16);
        acc = wmma_bf16(aq[kc], bk, acc);
      }
#pragma unroll
      for (int i = 0; i < 8; ++i) {
        int m = half ? i + 8 : i;
        Lf[m][colBase + l16] = acc[i] * scale;
      }
    }
  }
  __syncthreads();

  // ---------------- Phase 2: softmax per row, write attention ----------------
  {
    int row = tid >> 4;
    int sub = tid & 15;
    float* Lrow = Lf[row];
    float lm = -3.0e38f;
    for (int j = sub; j < S_LEN; j += 16) lm = fmaxf(lm, Lrow[j]);
#pragma unroll
    for (int off = 8; off >= 1; off >>= 1) lm = fmaxf(lm, __shfl_xor(lm, off, 16));
    float ls = 0.0f;
    for (int j = sub; j < S_LEN; j += 16) {
      float e = __expf(Lrow[j] - lm);
      Lrow[j] = e;
      ls += e;
    }
#pragma unroll
    for (int off = 8; off >= 1; off >>= 1) ls += __shfl_xor(ls, off, 16);
    float inv = 1.0f / ls;
    size_t attBase = ((size_t)bh * S_LEN + (qBase + row)) * S_LEN;
    for (int j = sub; j < S_LEN; j += 16) {
      float p = Lrow[j] * inv;
      Lrow[j] = p;
      att[attBase + j] = p;
    }
  }
  __syncthreads();

  // ---------------- Phase 3: values = A @ V (async V staging) ----------------
  {
    const int ntile = wid & 3;          // 4 tiles cover hd=64
    const int khalf = wid >> 2;         // K=32 half within each 64-row super-chunk
    const int ncol  = ntile * 16 + l16;
    const __bf16* vcol0 = qkvb + rowOff + hh * 192 + 128;   // V[k][n] base
    v8f acc = {};

    for (int sc = 0; sc < 32; ++sc) {
      const int kBase0 = sc * 64;
      // stage V[kBase0 .. kBase0+63][0..63] -> Vs via async LDS copies
#pragma unroll
      for (int j = 0; j < 2; ++j) {
        int s  = tid * 2 + j;           // 0..511 segments of 8 bf16
        int r  = s >> 3;
        int c8 = (s & 7) * 8;
        async_copy_b128(vcol0 + (size_t)(kBase0 + r) * QKV_ROW + c8, &Vs[r][c8]);
      }
      wait_async0();
      __syncthreads();

      // A fragment: rows = q rows, K slice from Lf (fp32 -> bf16)
      v16bf a;
      int kb = kBase0 + khalf * 32 + half * 16;
#pragma unroll
      for (int i = 0; i < 8; ++i) {
        a[i]     = (__bf16)Lf[l16][kb + i];
        a[i + 8] = (__bf16)Lf[l16][kb + 8 + i];
      }
      // B fragment: column ncol of Vs, K contiguous rows
      v16bf bv;
      int k0 = khalf * 32 + half * 16;
#pragma unroll
      for (int j = 0; j < 16; ++j) bv[j] = Vs[k0 + j][ncol];
      acc = wmma_bf16(a, bv, acc);
      __syncthreads();                  // Vs reused next iteration
    }

    // cross-wave K reduction (wave pairs w and w+4)
    if (khalf == 1) {
#pragma unroll
      for (int i = 0; i < 8; ++i) red3[ntile][half ? i + 8 : i][l16] = acc[i];
    }
    __syncthreads();
    if (khalf == 0) {
#pragma unroll
      for (int i = 0; i < 8; ++i) {
        int m = half ? i + 8 : i;
        float v = acc[i] + red3[ntile][m][l16];
        int s = qBase + m;
        vals[((size_t)(b * S_LEN + s)) * D_MODEL + hh * HD + ncol] = (__bf16)v;
      }
    }
  }
}

// ---------------------------------------------------------------------------
extern "C" void kernel_launch(void* const* d_in, const int* in_sizes, int n_in,
                              void* d_out, int out_size, void* d_ws, size_t ws_size,
                              hipStream_t stream) {
  const float* x    = (const float*)d_in[0];
  const float* Wqkv = (const float*)d_in[1];
  const float* bqkv = (const float*)d_in[2];
  const float* Wo   = (const float*)d_in[3];
  const float* bo   = (const float*)d_in[4];

  float* out = (float*)d_out;
  float* att = out + (size_t)2 * S_LEN * D_MODEL;

  __bf16* qkvb = (__bf16*)d_ws;
  __bf16* vals = (__bf16*)((char*)d_ws +
                           (size_t)2 * S_LEN * QKV_ROW * sizeof(__bf16));

  dim3 blk(256);
  gemm_bias<float, __bf16><<<dim3(QKV_ROW / 128, (2 * S_LEN) / 64), blk, 0, stream>>>(
      x, Wqkv, bqkv, qkvb, 2 * S_LEN, QKV_ROW, D_MODEL);
  attn_kernel<<<dim3(S_LEN / BQ, 2 * NH), blk, 0, stream>>>(qkvb, att, vals);
  gemm_bias<__bf16, float><<<dim3(D_MODEL / 128, (2 * S_LEN) / 64), blk, 0, stream>>>(
      vals, Wo, bo, out, 2 * S_LEN, D_MODEL, D_MODEL);
}
